// MyLSTM_89610197664434
// MI455X (gfx1250) — compile-verified, matrix operand order checked by reading, not run on hardware
//
#include <hip/hip_runtime.h>

typedef __attribute__((ext_vector_type(8)))  float  v8f;
typedef __attribute__((ext_vector_type(16))) __bf16 v16bf;
typedef __attribute__((ext_vector_type(8)))  __bf16 v8bf;

static constexpr int B = 64, S = 512, I = 512, H = 1024, G = 4 * H; // G = 4096
static constexpr int BH = B * H;                                    // 65536 elements

// ---------------------------------------------------------------------------
// Fragment load helper: one row of a row-major bf16 matrix, CDNA5 16-bit A/B
// layout. Lane L (0..15) holds K = {kb..kb+7, kb+16..kb+23}; lane L+16 holds
// K = {kb+8..kb+15, kb+24..kb+31}. Two 16-byte loads -> global_load_b128.
// ---------------------------------------------------------------------------
__device__ __forceinline__ v16bf ld_frag(const __bf16* __restrict__ p, int kLo) {
    v8bf lo = *(const v8bf*)(p + kLo);
    v8bf hi = *(const v8bf*)(p + kLo + 16);
    return __builtin_shufflevector(lo, hi, 0,1,2,3,4,5,6,7,8,9,10,11,12,13,14,15);
}

__device__ __forceinline__ float sigm(float x) { return 1.0f / (1.0f + __expf(-x)); }

// ---------------------------------------------------------------------------
// Fused LSTM cell for one timestep of one layer.
//   gates[64 x 4096] = [Ax | Ah] (64 x (D+1024)) @ W^T  + bias
// Block: 64 batch rows x 64 hidden cols x all 4 gates. 8 waves:
//   wave w: mi = w&3 (16 rows), nh = w>>2 (32 hidden cols = 2 N-tiles).
// Each wave: 8 accumulators (4 gates x 2 N-tiles), v_wmma_f32_16x16x32_bf16.
// Grid: H/64 = 16 blocks, 256 threads.
// ---------------------------------------------------------------------------
template <int D>
__global__ __launch_bounds__(256) void lstm_cell(
    const __bf16* __restrict__ Ax,   // (64, D)    x_t (layer0) or h0_cur (layer1)
    const __bf16* __restrict__ Ah,   // (64, 1024) h_prev of this layer (read buf)
    const __bf16* __restrict__ W,    // (4096, D+1024) packed [Wx|Wh], bf16
    const float*  __restrict__ bias, // (4096)  bx+bh
    float*        __restrict__ cbuf, // (64, 1024) cell state, updated in place
    __bf16*       __restrict__ hOutB,// (64, 1024) h_new bf16 (write buf)
    float*        __restrict__ hOutF,// (64, 1024) h_new f32 (for h_n)
    float*        __restrict__ out,  // (B,S,H) or nullptr
    int t)
{
    constexpr int KT = D + 1024;

    const int lane = threadIdx.x & 31;
    const int w    = threadIdx.x >> 5;
    const int mi   = w & 3;        // M tile: rows [16*mi, 16*mi+16)
    const int nh   = w >> 2;       // 0/1: 32-col strip within block's 64 cols
    const int l15  = lane & 15;
    const int kLo  = (lane >> 4) * 8;       // K sub-offset per lane half
    const int hc0  = blockIdx.x * 64 + nh * 32;

    // A operand: row = batch index, same row in both lane halves
    const int arow = mi * 16 + l15;
    const __bf16* aX = Ax + (size_t)arow * D;
    const __bf16* aH = Ah + (size_t)arow * 1024;

    // B operand: column n of W^T == row n of W (row-major, stride KT)
    const __bf16* wr[4][2];
    v8f acc[4][2];
#pragma unroll
    for (int g = 0; g < 4; ++g)
#pragma unroll
        for (int j = 0; j < 2; ++j) {
            const int n = g * 1024 + hc0 + j * 16 + l15;
            wr[g][j] = W + (size_t)n * KT;
            const float bv = bias[n];   // C layout: col == lane&15 for all r
#pragma unroll
            for (int r = 0; r < 8; ++r) acc[g][j][r] = bv;
        }

    // --- K loop part 1: input contribution (x_t @ Wx^T) ---
    for (int kb = 0; kb < D; kb += 32) {
        v16bf a = ld_frag(aX + kb, kLo);
#pragma unroll
        for (int g = 0; g < 4; ++g)
#pragma unroll
            for (int j = 0; j < 2; ++j) {
                v16bf b = ld_frag(wr[g][j] + kb, kLo);
                acc[g][j] = __builtin_amdgcn_wmma_f32_16x16x32_bf16(
                    false, a, false, b, (short)0, acc[g][j], false, false);
            }
    }
    // --- K loop part 2: recurrent contribution (h_prev @ Wh^T) ---
    for (int kb = 0; kb < 1024; kb += 32) {
        v16bf a = ld_frag(aH + kb, kLo);
#pragma unroll
        for (int g = 0; g < 4; ++g)
#pragma unroll
            for (int j = 0; j < 2; ++j) {
                v16bf b = ld_frag(wr[g][j] + D + kb, kLo);
                acc[g][j] = __builtin_amdgcn_wmma_f32_16x16x32_bf16(
                    false, a, false, b, (short)0, acc[g][j], false, false);
            }
    }

    // --- Elementwise gating, in-register on the accumulators ---
    // C/D layout: element r of v8f -> row (r + 8*(lane>>4)), col (lane&15)
#pragma unroll
    for (int j = 0; j < 2; ++j) {
        const int col = hc0 + j * 16 + l15;
#pragma unroll
        for (int r = 0; r < 8; ++r) {
            const int m   = mi * 16 + (lane >> 4) * 8 + r;
            const size_t idx = (size_t)m * 1024 + col;
            const float iv = sigm(acc[0][j][r]);
            const float fv = sigm(acc[1][j][r]);
            const float ov = sigm(acc[2][j][r]);
            const float gv = tanhf(acc[3][j][r]);
            const float cn = fv * cbuf[idx] + iv * gv;
            const float hn = ov * tanhf(cn);
            cbuf[idx]  = cn;
            hOutB[idx] = (__bf16)hn;
            hOutF[idx] = hn;
            if (out) out[((size_t)m * S + t) * 1024 + col] = hn;
        }
    }
}

// ---------------------------------------------------------------------------
// Prep kernels
// ---------------------------------------------------------------------------
// Pack [Wx | Wh] (fp32) -> bf16 row-major (4096, D+1024)
__global__ void pack_weights(const float* __restrict__ Wx, const float* __restrict__ Wh,
                             __bf16* __restrict__ outw, int D, long total) {
    const int KT = D + 1024;
    for (long i = (long)blockIdx.x * blockDim.x + threadIdx.x; i < total;
         i += (long)gridDim.x * blockDim.x) {
        const int k = (int)(i % KT);
        const int n = (int)(i / KT);
        const float v = (k < D) ? Wx[(size_t)n * D + k] : Wh[(size_t)n * 1024 + (k - D)];
        outw[i] = (__bf16)v;
    }
}

__global__ void fold_bias(const float* __restrict__ bx, const float* __restrict__ bh,
                          float* __restrict__ bo) {
    const int i = blockIdx.x * blockDim.x + threadIdx.x;
    if (i < G) bo[i] = bx[i] + bh[i];
}

// x (B,S,I) fp32 -> xb (S,B,I) bf16
__global__ void pack_x(const float* __restrict__ x, __bf16* __restrict__ xb) {
    const long total = (long)B * S * I;
    for (long i = (long)blockIdx.x * blockDim.x + threadIdx.x; i < total;
         i += (long)gridDim.x * blockDim.x) {
        const int ii = (int)(i % I);
        const long r = i / I;
        const int b  = (int)(r % B);
        const int s  = (int)(r / B);
        xb[i] = (__bf16)x[((size_t)b * S + s) * I + ii];
    }
}

__global__ void zero_words(unsigned* __restrict__ p, long nwords) {
    for (long i = (long)blockIdx.x * blockDim.x + threadIdx.x; i < nwords;
         i += (long)gridDim.x * blockDim.x)
        p[i] = 0u;
}

// Copy h_n / c_n into the tail of d_out: [h0, h1, c0, c1] each (B,H)
__global__ void copy_tail(const float* __restrict__ hf0, const float* __restrict__ hf1,
                          const float* __restrict__ c0,  const float* __restrict__ c1,
                          float* __restrict__ tail) {
    const int i = blockIdx.x * blockDim.x + threadIdx.x;
    if (i < BH) {
        tail[i]          = hf0[i];
        tail[BH + i]     = hf1[i];
        tail[2 * BH + i] = c0[i];
        tail[3 * BH + i] = c1[i];
    }
}

// ---------------------------------------------------------------------------
// Host launch
// ---------------------------------------------------------------------------
extern "C" void kernel_launch(void* const* d_in, const int* in_sizes, int n_in,
                              void* d_out, int out_size, void* d_ws, size_t ws_size,
                              hipStream_t stream) {
    const float* x   = (const float*)d_in[0];
    const float* Wx0 = (const float*)d_in[1];
    const float* Wh0 = (const float*)d_in[2];
    const float* bx0 = (const float*)d_in[3];
    const float* bh0 = (const float*)d_in[4];
    const float* Wx1 = (const float*)d_in[5];
    const float* Wh1 = (const float*)d_in[6];
    const float* bx1 = (const float*)d_in[7];
    const float* bh1 = (const float*)d_in[8];

    char* ws = (char*)d_ws;
    // Workspace layout (all offsets 256B-aligned)
    const size_t OFF_W0  = 0;                                     // (4096,1536) bf16
    const size_t OFF_W1  = OFF_W0 + (size_t)G * 1536 * 2;         // (4096,2048) bf16
    const size_t OFF_B0  = OFF_W1 + (size_t)G * 2048 * 2;         // 4096 f32
    const size_t OFF_B1  = OFF_B0 + (size_t)G * 4;
    const size_t OFF_XB  = OFF_B1 + (size_t)G * 4;                // (S,B,I) bf16
    const size_t OFF_ST  = OFF_XB + (size_t)S * B * I * 2;        // state region
    const size_t OFF_H0B = OFF_ST;                                // 2 x (B,H) bf16
    const size_t OFF_H1B = OFF_H0B + (size_t)2 * BH * 2;          // 2 x (B,H) bf16
    const size_t OFF_C0  = OFF_H1B + (size_t)2 * BH * 2;          // (B,H) f32
    const size_t OFF_C1  = OFF_C0 + (size_t)BH * 4;
    const size_t OFF_HF0 = OFF_C1 + (size_t)BH * 4;
    const size_t OFF_HF1 = OFF_HF0 + (size_t)BH * 4;
    const size_t STATE_BYTES = OFF_HF1 + (size_t)BH * 4 - OFF_ST;

    __bf16* wA0   = (__bf16*)(ws + OFF_W0);
    __bf16* wA1   = (__bf16*)(ws + OFF_W1);
    float*  bias0 = (float*)(ws + OFF_B0);
    float*  bias1 = (float*)(ws + OFF_B1);
    __bf16* xb    = (__bf16*)(ws + OFF_XB);
    __bf16* h0b   = (__bf16*)(ws + OFF_H0B);
    __bf16* h1b   = (__bf16*)(ws + OFF_H1B);
    float*  c0    = (float*)(ws + OFF_C0);
    float*  c1    = (float*)(ws + OFF_C1);
    float*  hf0   = (float*)(ws + OFF_HF0);
    float*  hf1   = (float*)(ws + OFF_HF1);
    float*  outp  = (float*)d_out;

    // ---- prep: weight/bias/x packing + state zero-init (deterministic) ----
    pack_weights<<<2048, 256, 0, stream>>>(Wx0, Wh0, wA0, 512,  (long)G * 1536);
    pack_weights<<<2048, 256, 0, stream>>>(Wx1, Wh1, wA1, 1024, (long)G * 2048);
    fold_bias<<<G / 256, 256, 0, stream>>>(bx0, bh0, bias0);
    fold_bias<<<G / 256, 256, 0, stream>>>(bx1, bh1, bias1);
    pack_x<<<8192, 256, 0, stream>>>(x, xb);
    zero_words<<<1024, 256, 0, stream>>>((unsigned*)(ws + OFF_ST), (long)(STATE_BYTES / 4));

    // ---- sequential timestep loop: 2 fused cell kernels per step ----
    for (int t = 0; t < S; ++t) {
        const int rd = t & 1, wrb = (t + 1) & 1;   // ping-pong for h
        const __bf16* xt  = xb + (size_t)t * B * I;
        const __bf16* h0r = h0b + (size_t)rd * BH;
        __bf16*       h0w = h0b + (size_t)wrb * BH;
        const __bf16* h1r = h1b + (size_t)rd * BH;
        __bf16*       h1w = h1b + (size_t)wrb * BH;

        lstm_cell<512><<<H / 64, 256, 0, stream>>>(
            xt, h0r, wA0, bias0, c0, h0w, hf0, nullptr, t);
        lstm_cell<1024><<<H / 64, 256, 0, stream>>>(
            h0w, h1r, wA1, bias1, c1, h1w, hf1, outp, t);
    }

    // ---- tail: h_n (2,B,H) then c_n (2,B,H) ----
    copy_tail<<<BH / 256, 256, 0, stream>>>(hf0, hf1, c0, c1,
                                            outp + (size_t)B * S * H);
}
